// LoRALinear_70360154243567
// MI455X (gfx1250) — matrix operand order repaired and checked by reading, not compiled, not run on hardware
//
#include <hip/hip_runtime.h>
#include <hip/hip_bf16.h>
#include <stddef.h>

// ---------------------------------------------------------------------------
// LoRA linear fused as single GEMM:
//   out = x @ (W + 2*B@A)^T + b
// 3xBF16 split-precision WMMA (hi*hi + hi*lo + lo*hi, f32 accumulate).
// ---------------------------------------------------------------------------

typedef __bf16 bf16_t;
typedef __attribute__((ext_vector_type(16))) __bf16 v16bf;
typedef __attribute__((ext_vector_type(8)))  __bf16 v8bf;
typedef __attribute__((ext_vector_type(8)))  float  v8f;

#define D_DIM   1024      // model dim (N == K)
#define R_LORA  16
#define SCALING 2.0f

#define BM   128          // block tile M
#define BN   128          // block tile N
#define BK   32           // K step (== WMMA K for bf16)
#define ROWE 40           // LDS row: 32 elems + 8 pad (80B stride, conflict-free)
#define NKT  (D_DIM / BK) // 32 K steps

union Frag {
    v16bf v;
    v8bf  h[2];
};

union F16x {             // 16 floats <-> 4x float4 for 64B global loads
    float4 q[4];
    float  f[16];
};

// ---- Kernel 1: Weff[o][d] = W[o][d] + 2 * sum_r B[o][r]*A[r][d] -----------
__global__ __launch_bounds__(256)
void build_weff_kernel(const float* __restrict__ W,
                       const float* __restrict__ A,
                       const float* __restrict__ B,
                       float* __restrict__ Weff) {
    const int d = blockIdx.x * 256 + threadIdx.x;
    const int o = blockIdx.y;
    float acc = 0.0f;
#pragma unroll
    for (int r = 0; r < R_LORA; ++r)
        acc += B[o * R_LORA + r] * A[r * D_DIM + d];
    Weff[(size_t)o * D_DIM + d] = W[(size_t)o * D_DIM + d] + SCALING * acc;
}

// ---- Kernel 2: out[m][n] = sum_k x[m][k]*Weff[n][k] + b[n] ----------------
__global__ __launch_bounds__(256)
void lora_gemm_kernel(const float* __restrict__ X,
                      const float* __restrict__ Wf,
                      const float* __restrict__ bias,
                      float* __restrict__ out) {
    // hi/lo bf16 planes, double buffered: 4 * 2 * 128*40*2B = 80 KB
    __shared__ alignas(16) bf16_t Xh[2][BM * ROWE];
    __shared__ alignas(16) bf16_t Xl[2][BM * ROWE];
    __shared__ alignas(16) bf16_t Wh[2][BN * ROWE];
    __shared__ alignas(16) bf16_t Wl[2][BN * ROWE];

    const int tid   = threadIdx.x;
    const int lane  = tid & 31;
    const int wave  = tid >> 5;       // 0..7
    const int waveM = wave >> 2;      // 0..1  (64-row strip)
    const int waveN = wave & 3;       // 0..3  (32-col strip)
    const int half  = lane >> 4;      // half-wave select
    const int l16   = lane & 15;

    const int mBlock = blockIdx.y * BM;
    const int nBlock = blockIdx.x * BN;

    // Global staging: 2 threads per tile row, 16 consecutive f32 (64B) each.
    const int gRow = tid >> 1;                 // 0..127
    const int gCol = (tid & 1) * 16;           // 0 or 16
    const float* xPtr = X  + (size_t)(mBlock + gRow) * D_DIM + gCol;
    const float* wPtr = Wf + (size_t)(nBlock + gRow) * D_DIM + gCol;

    F16x xr, wr;

    auto gload = [&](int kt) {
        const float4* px = reinterpret_cast<const float4*>(xPtr + kt * BK);
        const float4* pw = reinterpret_cast<const float4*>(wPtr + kt * BK);
#pragma unroll
        for (int i = 0; i < 4; ++i) { xr.q[i] = px[i]; wr.q[i] = pw[i]; }
    };

    auto stage = [&](int buf) {
        alignas(16) bf16_t xh[16], xl[16], wh[16], wl[16];
#pragma unroll
        for (int i = 0; i < 16; ++i) {
            float xf = xr.f[i];
            bf16_t hx = (bf16_t)xf;
            xh[i] = hx;
            xl[i] = (bf16_t)(xf - (float)hx);
            float wf = wr.f[i];
            bf16_t hw = (bf16_t)wf;
            wh[i] = hw;
            wl[i] = (bf16_t)(wf - (float)hw);
        }
        const int o = gRow * ROWE + gCol;
#pragma unroll
        for (int p = 0; p < 2; ++p) {
            *(v8bf*)&Xh[buf][o + 8 * p] = *(const v8bf*)&xh[8 * p];
            *(v8bf*)&Xl[buf][o + 8 * p] = *(const v8bf*)&xl[8 * p];
            *(v8bf*)&Wh[buf][o + 8 * p] = *(const v8bf*)&wh[8 * p];
            *(v8bf*)&Wl[buf][o + 8 * p] = *(const v8bf*)&wl[8 * p];
        }
    };

    v8f acc[4][2];
    const v8f vzero = {0.f, 0.f, 0.f, 0.f, 0.f, 0.f, 0.f, 0.f};
#pragma unroll
    for (int fm = 0; fm < 4; ++fm)
#pragma unroll
        for (int fn = 0; fn < 2; ++fn) acc[fm][fn] = vzero;

    auto compute = [&](int buf) {
        // A-fragments (x tile): lane holds row m; K runs [8h,8h+8) & [16+8h,+8)
        Frag ah[4], al[4];
#pragma unroll
        for (int fm = 0; fm < 4; ++fm) {
            const int m = waveM * 64 + fm * 16 + l16;
            const bf16_t* rh = &Xh[buf][m * ROWE];
            const bf16_t* rl = &Xl[buf][m * ROWE];
            ah[fm].h[0] = *(const v8bf*)(rh + 8 * half);
            ah[fm].h[1] = *(const v8bf*)(rh + 16 + 8 * half);
            al[fm].h[0] = *(const v8bf*)(rl + 8 * half);
            al[fm].h[1] = *(const v8bf*)(rl + 16 + 8 * half);
        }
        // B-fragments (Weff tile, K-major per lane): lane holds col n;
        // K runs [8h,8h+8) & [16+8h,+8).
        Frag bh[2], bl[2];
#pragma unroll
        for (int fn = 0; fn < 2; ++fn) {
            const int n = waveN * 32 + fn * 16 + l16;
            const bf16_t* rh = &Wh[buf][n * ROWE];
            const bf16_t* rl = &Wl[buf][n * ROWE];
            bh[fn].h[0] = *(const v8bf*)(rh + 8 * half);
            bh[fn].h[1] = *(const v8bf*)(rh + 16 + 8 * half);
            bl[fn].h[0] = *(const v8bf*)(rl + 8 * half);
            bl[fn].h[1] = *(const v8bf*)(rl + 16 + 8 * half);
        }
#pragma unroll
        for (int fm = 0; fm < 4; ++fm) {
#pragma unroll
            for (int fn = 0; fn < 2; ++fn) {
                v8f c = acc[fm][fn];
                c = __builtin_amdgcn_wmma_f32_16x16x32_bf16(
                        false, ah[fm].v, false, bh[fn].v, (short)0, c, false, false);
                c = __builtin_amdgcn_wmma_f32_16x16x32_bf16(
                        false, ah[fm].v, false, bl[fn].v, (short)0, c, false, false);
                c = __builtin_amdgcn_wmma_f32_16x16x32_bf16(
                        false, al[fm].v, false, bh[fn].v, (short)0, c, false, false);
                acc[fm][fn] = c;
            }
        }
    };

    // ---- pipeline: register-prefetch next global tile, double-buffer LDS ----
    gload(0);
    stage(0);
    __syncthreads();

    for (int kt = 0; kt < NKT; ++kt) {
        const int buf = kt & 1;
        if (kt + 1 < NKT) gload(kt + 1);
        compute(buf);
        if (kt + 1 < NKT) stage(buf ^ 1);
        __syncthreads();
    }

    // ---- epilogue: bias add (n is constant per lane across a C-frag) ----
    float bv[2];
#pragma unroll
    for (int fn = 0; fn < 2; ++fn)
        bv[fn] = bias[nBlock + waveN * 32 + fn * 16 + l16];

#pragma unroll
    for (int fm = 0; fm < 4; ++fm) {
#pragma unroll
        for (int fn = 0; fn < 2; ++fn) {
            const int n     = nBlock + waveN * 32 + fn * 16 + l16;
            const int mBase = mBlock + waveM * 64 + fm * 16 + 8 * half;
            v8f a = acc[fm][fn];
#pragma unroll
            for (int v = 0; v < 8; ++v)
                out[(size_t)(mBase + v) * D_DIM + n] = a[v] + bv[fn];
        }
    }
}

// ---------------------------------------------------------------------------
extern "C" void kernel_launch(void* const* d_in, const int* in_sizes, int n_in,
                              void* d_out, int out_size, void* d_ws, size_t ws_size,
                              hipStream_t stream) {
    const float* x = (const float*)d_in[0];   // [4,4096,1024]
    const float* W = (const float*)d_in[1];   // [1024,1024] (out,in)
    const float* b = (const float*)d_in[2];   // [1024]
    const float* A = (const float*)d_in[3];   // [16,1024]
    const float* B = (const float*)d_in[4];   // [1024,16]
    float* out  = (float*)d_out;
    float* Weff = (float*)d_ws;               // 1024*1024*4 = 4 MB scratch

    const int M = in_sizes[0] / D_DIM;        // 16384

    // 1) Weff = W + 2*B@A
    build_weff_kernel<<<dim3(D_DIM / 256, D_DIM), 256, 0, stream>>>(W, A, B, Weff);

    // 2) out = x @ Weff^T + b   (128x128 tiles, 8 N-tiles x 128 M-tiles)
    lora_gemm_kernel<<<dim3(D_DIM / BN, M / BM), 256, 0, stream>>>(x, Weff, b, out);
}
